// Classify10_74242804678792
// MI455X (gfx1250) — matrix operand order repaired and checked by reading, not compiled, not run on hardware
//
#include <hip/hip_runtime.h>
#include <hip/hip_bf16.h>

// ---------------------------------------------------------------------------
// MI455X (gfx1250) fused transformer-layer classifier, WMMA f16 path.
// All fragment traffic is wide: A-frags = 2 x b128/lane, weight B-frags are
// pre-packed into fragment order (1 x 32B contiguous load/lane, coalesced),
// K^T and V^T B-frags are contiguous v16h loads. Workspace ~349 MB.
// ---------------------------------------------------------------------------

typedef _Float16 half_t;
typedef __attribute__((ext_vector_type(16))) _Float16 v16h;
typedef __attribute__((ext_vector_type(8)))  _Float16 v8h;
typedef __attribute__((ext_vector_type(8)))  float    v8f;

#define BATCH 256
#define SEQ   360
#define SP    384           // padded sequence (24 tiles of 16)
#define DIM   256
#define HEADS 4
#define DK    64
#define DFF   2048
#define PREFIX_LEN 340
#define PADTOK 799
#define NPAIR 91

__device__ __forceinline__ v8f wmma16(v16h a, v16h b, v8f c) {
  return __builtin_amdgcn_wmma_f32_16x16x32_f16(false, a, false, b, (short)0, c, false, false);
}

// A fragment (16x32 tile, row-major, lda elems): lane m=lane&15, hi=lane>>4,
// holds k = hi*16 + e (e<16)  -> two contiguous v8h.
__device__ __forceinline__ v16h load_a(const half_t* A, int lda) {
  int lane = threadIdx.x & 31;
  int hi = lane >> 4, m = lane & 15;
  const half_t* row = A + (size_t)m * lda + hi * 16;
  v8h lo = *(const v8h*)(row);
  v8h hh = *(const v8h*)(row + 8);
  return __builtin_shufflevector(lo, hh, 0, 1, 2, 3, 4, 5, 6, 7,
                                 8, 9, 10, 11, 12, 13, 14, 15);
}

// B fragment from pre-packed weights: lane's 16 values contiguous (32 B).
__device__ __forceinline__ v16h load_bp(const half_t* Wp) {
  int lane = threadIdx.x & 31;
  return *(const v16h*)(Wp + (size_t)lane * 16);
}

// B fragment with element (k,n) = M[n*ldm + k]; k = hi*16 + e is contiguous.
// Used for K^T (rows of K) and V^T (rows of vT).
__device__ __forceinline__ v16h load_bT(const half_t* M, int ldm) {
  int lane = threadIdx.x & 31;
  int hi = lane >> 4, n = lane & 15;
  return *(const v16h*)(M + (size_t)n * ldm + hi * 16);
}

// ---------------------------------------------------------------------------
// Pack f32 weight (K x N row-major) into WMMA-B fragment order:
// out[((kt*(N/16)+nt)*32 + lane)*16 + e] = W[(kt*32 + (lane>>4)*16 + e)*N + nt*16 + (lane&15)]
__global__ void pack_w_kernel(const float* __restrict__ W, half_t* __restrict__ out, int N) {
  int i = blockIdx.x * 256 + threadIdx.x;        // exact multiple launches only
  int e = i & 15, lane = (i >> 4) & 31, t = i >> 9;
  int Nt = N >> 4;
  int nt = t % Nt, kt = t / Nt;
  int hi = lane >> 4, n = lane & 15;
  int k = kt * 32 + hi * 16 + e;
  out[i] = (half_t)W[(size_t)k * N + nt * 16 + n];
}

// Embedding + sqrt(D)*scale + sinusoidal PE on the prefix -> x (f32)
__global__ void embed_kernel(const int* __restrict__ src, const float* __restrict__ emb,
                             float* __restrict__ x) {
  int bs = blockIdx.x;                 // b*SEQ + s
  int b = bs / SEQ, s = bs % SEQ;
  int d = threadIdx.x;
  int tok = src[b * SEQ + s];
  float v = emb[(size_t)tok * DIM + d];
  if (s < PREFIX_LEN) {
    float ex = (float)d / 128.0f;                 // 2*d/D
    float theta = (float)s * powf(10000.0f, -ex);
    float pe = (d & 1) ? cosf(theta) : sinf(theta);
    v = v * 16.0f + pe;                           // sqrt(256)=16
  }
  x[((size_t)b * SEQ + s) * DIM + d] = v;
}

// LayerNorm (ddof=1) -> f16; padded rows (s>=SEQ) written as zero.
__global__ void ln_kernel(const float* __restrict__ x, half_t* __restrict__ out,
                          const float* __restrict__ ga, const float* __restrict__ be) {
  int bs = blockIdx.x;                 // b*SP + s
  int b = bs / SP, s = bs % SP;
  int d = threadIdx.x;
  size_t orow = (size_t)bs * DIM;
  if (s >= SEQ) { out[orow + d] = (half_t)0.0f; return; }
  float v = x[((size_t)b * SEQ + s) * DIM + d];
  __shared__ float red[256];
  red[d] = v; __syncthreads();
  for (int off = 128; off > 0; off >>= 1) { if (d < off) red[d] += red[d + off]; __syncthreads(); }
  float mu = red[0] * (1.0f / 256.0f);
  __syncthreads();
  float dv = v - mu;
  red[d] = dv * dv; __syncthreads();
  for (int off = 128; off > 0; off >>= 1) { if (d < off) red[d] += red[d + off]; __syncthreads(); }
  float sd = sqrtf(red[0] * (1.0f / 255.0f));     // ddof=1
  out[orow + d] = (half_t)(ga[d] * dv / (sd + 1e-6f) + be[d]);
}

// C(SPx256,f16) = A(SPx256,f16) @ Wp(packed 256x256) + bias.
// One m-tile per block row; each wave: 2 n-tiles, A-frags hoisted.
__global__ void gemm_qkv_kernel(const half_t* __restrict__ A, const half_t* __restrict__ Wp,
                                const float* __restrict__ bias, half_t* __restrict__ out) {
  int b = blockIdx.x, mt = blockIdx.y;           // mt 0..23
  int wave = threadIdx.x >> 5, lane = threadIdx.x & 31, hi = lane >> 4, ln = lane & 15;
  const half_t* Ab = A + (size_t)b * SP * DIM + (size_t)mt * 16 * DIM;
  v16h af[8];
#pragma unroll
  for (int ks = 0; ks < 8; ++ks) af[ks] = load_a(Ab + ks * 32, DIM);
#pragma unroll
  for (int t = 0; t < 2; ++t) {
    int nt = wave + t * 8;                        // 0..15
    v8f acc = {};
#pragma unroll
    for (int ks = 0; ks < 8; ++ks)
      acc = wmma16(af[ks], load_bp(Wp + (size_t)(ks * 16 + nt) * 512), acc);
    float bv = bias[nt * 16 + ln];
    half_t* ob = out + (size_t)b * SP * DIM + nt * 16 + ln;
#pragma unroll
    for (int r = 0; r < 8; ++r)
      ob[(size_t)(mt * 16 + r + 8 * hi) * DIM] = (half_t)(acc[r] + bv);
  }
}

// Same GEMM but stores V TRANSPOSED: vT[b][n][m], ld = SP.
// The 8 C rows per lane are contiguous in m -> one v8h store.
__global__ void gemm_qkv_t_kernel(const half_t* __restrict__ A, const half_t* __restrict__ Wp,
                                  const float* __restrict__ bias, half_t* __restrict__ vT) {
  int b = blockIdx.x, mt = blockIdx.y;           // mt 0..23
  int wave = threadIdx.x >> 5, lane = threadIdx.x & 31, hi = lane >> 4, ln = lane & 15;
  const half_t* Ab = A + (size_t)b * SP * DIM + (size_t)mt * 16 * DIM;
  v16h af[8];
#pragma unroll
  for (int ks = 0; ks < 8; ++ks) af[ks] = load_a(Ab + ks * 32, DIM);
#pragma unroll
  for (int t = 0; t < 2; ++t) {
    int nt = wave + t * 8;
    v8f acc = {};
#pragma unroll
    for (int ks = 0; ks < 8; ++ks)
      acc = wmma16(af[ks], load_bp(Wp + (size_t)(ks * 16 + nt) * 512), acc);
    float bv = bias[nt * 16 + ln];
    v8h st;
#pragma unroll
    for (int r = 0; r < 8; ++r) st[r] = (half_t)(acc[r] + bv);
    *(v8h*)(vT + (size_t)b * DIM * SP + (size_t)(nt * 16 + ln) * SP + mt * 16 + 8 * hi) = st;
  }
}

// x(360x256,f32) += ctx(f16) @ Wo(packed) + bo    (23 m-tiles, guard m<360)
__global__ void gemm_oproj_kernel(const half_t* __restrict__ A, const half_t* __restrict__ Wp,
                                  const float* __restrict__ bias, float* __restrict__ x) {
  int b = blockIdx.x, mt = blockIdx.y;           // mt 0..22
  int wave = threadIdx.x >> 5, lane = threadIdx.x & 31, hi = lane >> 4, ln = lane & 15;
  const half_t* Ab = A + (size_t)b * SP * DIM + (size_t)mt * 16 * DIM;
  v16h af[8];
#pragma unroll
  for (int ks = 0; ks < 8; ++ks) af[ks] = load_a(Ab + ks * 32, DIM);
#pragma unroll
  for (int t = 0; t < 2; ++t) {
    int nt = wave + t * 8;
    v8f acc = {};
#pragma unroll
    for (int ks = 0; ks < 8; ++ks)
      acc = wmma16(af[ks], load_bp(Wp + (size_t)(ks * 16 + nt) * 512), acc);
    float bv = bias[nt * 16 + ln];
    float* xb = x + (size_t)b * SEQ * DIM + nt * 16 + ln;
#pragma unroll
    for (int r = 0; r < 8; ++r) {
      int m = mt * 16 + r + 8 * hi;
      if (m < SEQ) xb[(size_t)m * DIM] += acc[r] + bv;
    }
  }
}

// Attention for one (b, h): scores via WMMA (Q K^T, K^T frag contiguous),
// softmax with the reference's QUERY-side pad mask (padded query -> uniform),
// probs -> f16 in LDS, then P @ V^T via WMMA (V^T frag contiguous).
__global__ void attn_kernel(const int* __restrict__ src,
                            const half_t* __restrict__ q, const half_t* __restrict__ k,
                            const half_t* __restrict__ vT, half_t* __restrict__ ctx) {
  __shared__ __align__(32) float  sc[16][SP];    // 24 KB
  __shared__ __align__(32) half_t pf[16][SP];    // 12 KB
  int b = blockIdx.x, h = blockIdx.y;
  int tid = threadIdx.x, wave = tid >> 5, lane = tid & 31, hi = lane >> 4, ln = lane & 15;
  const half_t* qb  = q  + (size_t)b * SP * DIM + h * DK;
  const half_t* kb  = k  + (size_t)b * SP * DIM + h * DK;
  const half_t* vTb = vT + (size_t)b * DIM * SP + (size_t)h * DK * SP;
  half_t* cb = ctx + (size_t)b * SP * DIM + h * DK;
  const int* sb = src + (size_t)b * SEQ;

  for (int qt = 0; qt < 23; ++qt) {    // cover rows 0..367 (>=360 junk, unused)
    // ---- Q-tile A fragments (hoisted): DK=64 -> 2 k-steps
    v16h qa[2];
#pragma unroll
    for (int ks = 0; ks < 2; ++ks) qa[ks] = load_a(qb + (size_t)(qt * 16) * DIM + ks * 32, DIM);

    // ---- scores: 24 key tiles split across 8 waves
    for (int kt = wave; kt < 24; kt += 8) {
      v8f acc = {};
#pragma unroll
      for (int ks = 0; ks < 2; ++ks)
        acc = wmma16(qa[ks], load_bT(kb + (size_t)(kt * 16) * DIM + ks * 32, DIM), acc);
#pragma unroll
      for (int r = 0; r < 8; ++r)
        sc[r + 8 * hi][kt * 16 + ln] = acc[r] * 0.125f;   // 1/sqrt(64)
    }
    __syncthreads();

    // ---- softmax: 16 threads per row (contiguous half-wave), 24 cols each
    {
      int row = tid >> 4, cl = tid & 15;
      int m = qt * 16 + row;
      bool qpad = (m >= SEQ) || (sb[m] == PADTOK);
      float mx = -1e30f;
      for (int i = 0; i < 24; ++i) {
        int c = cl + 16 * i;
        if (c < SEQ) mx = fmaxf(mx, sc[row][c]);
      }
      for (int o = 8; o > 0; o >>= 1) mx = fmaxf(mx, __shfl_xor(mx, o, 16));
      float sum = 0.0f;
      for (int i = 0; i < 24; ++i) {
        int c = cl + 16 * i;
        if (c < SEQ) sum += __expf(sc[row][c] - mx);
      }
      for (int o = 8; o > 0; o >>= 1) sum += __shfl_xor(sum, o, 16);
      float inv = 1.0f / sum;
      for (int i = 0; i < 24; ++i) {
        int c = cl + 16 * i;
        float p;
        if (c >= SEQ) p = 0.0f;
        else if (qpad) p = 1.0f / 360.0f;        // uniform (all scores -1e9)
        else p = __expf(sc[row][c] - mx) * inv;
        pf[row][c] = (half_t)p;
      }
    }
    __syncthreads();

    // ---- P @ V^T : waves 0..3, one 16x16 ctx tile each; 12 k-steps (384 keys)
    if (wave < 4) {
      int nt = wave;
      v8f acc = {};
      for (int ks = 0; ks < 12; ++ks) {
        v16h a = load_a(&pf[0][0] + ks * 32, SP);
        v16h bb = load_bT(vTb + (size_t)(nt * 16) * SP + ks * 32, SP);
        acc = wmma16(a, bb, acc);
      }
#pragma unroll
      for (int r = 0; r < 8; ++r)
        cb[(size_t)(qt * 16 + r + 8 * hi) * DIM + nt * 16 + ln] = (half_t)acc[r];
    }
    __syncthreads();
  }
}

// Fused FFN per 16-row tile: h = relu(xn2@W1+b1) kept in 64KB LDS (f16),
// then x += h@W2 + b2.  W1/W2 pre-packed.
__global__ void ffn_kernel(const half_t* __restrict__ xn2,
                           const half_t* __restrict__ w1p, const float* __restrict__ b1,
                           const half_t* __restrict__ w2p, const float* __restrict__ b2,
                           float* __restrict__ x) {
  extern __shared__ __align__(128) half_t hbuf[];   // 16 x 2048 f16 = 65536 B
  int b = blockIdx.x, mt = blockIdx.y;              // mt 0..22
  int tid = threadIdx.x, wave = tid >> 5, lane = tid & 31, hi = lane >> 4, ln = lane & 15;
  const half_t* Ab = xn2 + (size_t)b * SP * DIM + (size_t)mt * 16 * DIM;

  v16h af[8];
#pragma unroll
  for (int ks = 0; ks < 8; ++ks) af[ks] = load_a(Ab + ks * 32, DIM);

  // stage 1: 128 hidden n-tiles; packed W1 tile offset = (ks*128 + nt)*512
  for (int nt = wave; nt < 128; nt += 8) {
    v8f acc = {};
#pragma unroll
    for (int ks = 0; ks < 8; ++ks)
      acc = wmma16(af[ks], load_bp(w1p + (size_t)(ks * 128 + nt) * 512), acc);
    float bv = b1[nt * 16 + ln];
#pragma unroll
    for (int r = 0; r < 8; ++r) {
      float t = fmaxf(acc[r] + bv, 0.0f);           // relu
      hbuf[(size_t)(r + 8 * hi) * DFF + nt * 16 + ln] = (half_t)t;
    }
  }
  __syncthreads();

  // stage 2: 16 output n-tiles, K=2048 -> 64 k-steps from LDS
  for (int nt = wave; nt < 16; nt += 8) {
    v8f acc = {};
    for (int ks = 0; ks < 64; ++ks) {
      v16h a = load_a(hbuf + ks * 32, DFF);
      acc = wmma16(a, load_bp(w2p + (size_t)(ks * 16 + nt) * 512), acc);
    }
    float bv = b2[nt * 16 + ln];
    float* xb = x + (size_t)b * SEQ * DIM + nt * 16 + ln;
#pragma unroll
    for (int r = 0; r < 8; ++r) {
      int m = mt * 16 + r + 8 * hi;
      if (m < SEQ) xb[(size_t)m * DIM] += acc[r] + bv;
    }
  }
}

// Pair-feature gather -> 2048-dot with wl (3 outputs) + output masking.
__global__ void logits_kernel(const int* __restrict__ src, const float* __restrict__ x,
                              const float* __restrict__ wl, const float* __restrict__ bl,
                              float* __restrict__ out) {
  int b = blockIdx.x;
  int tid = threadIdx.x, wave = tid >> 5, lane = tid & 31;
  const float* xb = x + (size_t)b * SEQ * DIM;
  const int* sb = src + (size_t)b * SEQ;
  for (int p = wave; p < NPAIR; p += 8) {
    int i = 0, j = 0, rem = p;
    for (int ii = 0; ii < 13; ++ii) {
      int cnt = 13 - ii;
      if (rem < cnt) { i = ii; j = ii + 1 + rem; break; }
      rem -= cnt;
    }
    int pei = 12 * i - (i * (i - 1)) / 2 + (j - i) - 1;
    int eb = 28 + 2 * pei;
    int gidx[8] = {2 * i, 2 * i + 1, 2 * j, 2 * j + 1, eb, eb + 1, eb + 2, eb + 3};
    float a0 = 0.0f, a1 = 0.0f, a2 = 0.0f;
#pragma unroll
    for (int g = 0; g < 8; ++g) {
      float zm = (g >= 4 && j == 13) ? 0.0f : 1.0f;   // ZMASK
      const float* row = xb + (size_t)gidx[g] * DIM;
      for (int d = lane; d < DIM; d += 32) {
        float f = row[d] * zm;
        const float* wr = wl + (size_t)(g * DIM + d) * 3;
        a0 += f * wr[0]; a1 += f * wr[1]; a2 += f * wr[2];
      }
    }
    for (int o = 16; o > 0; o >>= 1) {
      a0 += __shfl_xor(a0, o, 32);
      a1 += __shfl_xor(a1, o, 32);
      a2 += __shfl_xor(a2, o, 32);
    }
    if (lane == 0) {
      int ti = sb[2 * i], ai = sb[2 * i + 1], tj = sb[2 * j], aj = sb[2 * j + 1];
      bool c1 = (ai == 2) || (aj == 2);
      bool c2 = (tj == 1) || (ai == 1) || (aj == 1);
      bool c3 = (ti == PADTOK) || (tj == PADTOK) || (ai == 0) || (aj == 0);
      bool c4 = false;
      if (j < 13) {
        int es = 28 + 4 * pei;
#pragma unroll
        for (int qq = 0; qq < 4; ++qq) c4 |= (sb[es + qq] == 1);
      }
      bool m111 = c3 || c4;
      float* op = out + ((size_t)b * NPAIR + p) * 3;
      op[0] = m111 ? -1e9f : (a0 + bl[0]);
      op[1] = (m111 || c2) ? -1e9f : (a1 + bl[1]);
      op[2] = (m111 || c2 || c1) ? -1e9f : (a2 + bl[2]);
    }
  }
}

// ---------------------------------------------------------------------------
extern "C" void kernel_launch(void* const* d_in, const int* in_sizes, int n_in,
                              void* d_out, int out_size, void* d_ws, size_t ws_size,
                              hipStream_t stream) {
  const int*   src = (const int*)  d_in[0];
  const float* emb = (const float*)d_in[1];
  const float* wq  = (const float*)d_in[2];  const float* bq = (const float*)d_in[3];
  const float* wk  = (const float*)d_in[4];  const float* bk = (const float*)d_in[5];
  const float* wv  = (const float*)d_in[6];  const float* bv = (const float*)d_in[7];
  const float* wo  = (const float*)d_in[8];  const float* bo = (const float*)d_in[9];
  const float* n1a = (const float*)d_in[10]; const float* n1b = (const float*)d_in[11];
  const float* n2a = (const float*)d_in[12]; const float* n2b = (const float*)d_in[13];
  const float* w1  = (const float*)d_in[14]; const float* b1 = (const float*)d_in[15];
  const float* w2  = (const float*)d_in[16]; const float* b2 = (const float*)d_in[17];
  const float* wl  = (const float*)d_in[18]; const float* bl = (const float*)d_in[19];
  float* out = (float*)d_out;

  // workspace layout
  char* ws = (char*)d_ws;
  size_t off = 0;
  auto take = [&](size_t bytes) { char* p = ws + off; off += (bytes + 255) & ~size_t(255); return p; };
  float*  x    = (float*) take((size_t)BATCH * SEQ * DIM * 4);
  half_t* xn   = (half_t*)take((size_t)BATCH * SP  * DIM * 2);  // reused as xn2
  half_t* qb   = (half_t*)take((size_t)BATCH * SP  * DIM * 2);
  half_t* kb   = (half_t*)take((size_t)BATCH * SP  * DIM * 2);
  half_t* vTb  = (half_t*)take((size_t)BATCH * DIM * SP  * 2);  // V transposed
  half_t* ctx  = (half_t*)take((size_t)BATCH * SP  * DIM * 2);
  half_t* wqP  = (half_t*)take((size_t)DIM * DIM * 2);
  half_t* wkP  = (half_t*)take((size_t)DIM * DIM * 2);
  half_t* wvP  = (half_t*)take((size_t)DIM * DIM * 2);
  half_t* woP  = (half_t*)take((size_t)DIM * DIM * 2);
  half_t* w1P  = (half_t*)take((size_t)DIM * DFF * 2);
  half_t* w2P  = (half_t*)take((size_t)DFF * DIM * 2);
  (void)ws_size; (void)n_in; (void)in_sizes; (void)out_size;

  // 1) weights -> packed f16 fragment order
  pack_w_kernel<<<(DIM * DIM) / 256, 256, 0, stream>>>(wq, wqP, DIM);
  pack_w_kernel<<<(DIM * DIM) / 256, 256, 0, stream>>>(wk, wkP, DIM);
  pack_w_kernel<<<(DIM * DIM) / 256, 256, 0, stream>>>(wv, wvP, DIM);
  pack_w_kernel<<<(DIM * DIM) / 256, 256, 0, stream>>>(wo, woP, DIM);
  pack_w_kernel<<<(DIM * DFF) / 256, 256, 0, stream>>>(w1, w1P, DFF);
  pack_w_kernel<<<(DFF * DIM) / 256, 256, 0, stream>>>(w2, w2P, DIM);

  // 2) embed + PE
  embed_kernel<<<BATCH * SEQ, 256, 0, stream>>>(src, emb, x);
  // 3) LN1 -> xn (f16, padded rows zero)
  ln_kernel<<<BATCH * SP, 256, 0, stream>>>(x, xn, n1a, n1b);
  // 4) Q, K (row-major), V (transposed)
  gemm_qkv_kernel  <<<dim3(BATCH, 24), 256, 0, stream>>>(xn, wqP, bq, qb);
  gemm_qkv_kernel  <<<dim3(BATCH, 24), 256, 0, stream>>>(xn, wkP, bk, kb);
  gemm_qkv_t_kernel<<<dim3(BATCH, 24), 256, 0, stream>>>(xn, wvP, bv, vTb);
  // 5) attention
  attn_kernel<<<dim3(BATCH, HEADS), 256, 0, stream>>>(src, qb, kb, vTb, ctx);
  // 6) output projection + residual
  gemm_oproj_kernel<<<dim3(BATCH, 23), 256, 0, stream>>>(ctx, woP, bo, x);
  // 7) LN2 -> xn (reuse buffer)
  ln_kernel<<<BATCH * SP, 256, 0, stream>>>(x, xn, n2a, n2b);
  // 8) fused FFN (+residual into x); 64 KB dynamic LDS hidden tile
  ffn_kernel<<<dim3(BATCH, 23), 256, 16 * DFF * sizeof(half_t), stream>>>(xn, w1P, b1, w2P, b2, x);
  // 9) gather + logits + masking
  logits_kernel<<<BATCH, 256, 0, stream>>>(src, x, wl, bl, out);
}
